// SparseTransformer_38594576122323
// MI455X (gfx1250) — matrix operand order repaired and checked by reading, not compiled, not run on hardware
//
#include <hip/hip_runtime.h>

// ---------------------------------------------------------------------------
// MI455X (gfx1250) sparse-transformer forward, f16 WMMA pipeline.
//  * all activations/weights converted to f16 once per launch; weights also
//    pre-transposed to [N][K] so every LDS fragment gather is contiguous
//    (2x ds_load_b128 + shufflevector concat, no per-half packing)
//  * tile staging uses gfx1250 GLOBAL_LOAD_ASYNC_TO_LDS_B128 (ASYNCcnt) —
//    builtin confirmed present; its params are `int4*` (generic), per probe
//  * exact tiling everywhere (M%128, N%64, K%32 == 0) -> no bounds branches
// ---------------------------------------------------------------------------

#define D_MODEL 512
#define N_HEADS 8
#define DEPTH 64
#define DFF 2048
#define HIST 1024
#define FLEN 256
#define FEAT 64
#define BATCH 4
#define ENC_BLOCKS 8

typedef __attribute__((ext_vector_type(16))) _Float16 v16h;
typedef __attribute__((ext_vector_type(8)))  _Float16 v8h;
typedef __attribute__((ext_vector_type(8)))  float    v8f;
typedef int v4i __attribute__((vector_size(16)));

#if defined(__gfx1250__) && __has_builtin(__builtin_amdgcn_global_load_async_to_lds_b128) && __has_builtin(__builtin_amdgcn_s_wait_asynccnt)
#define USE_ASYNC 1
#else
#define USE_ASYNC 0
#endif

// copy 8 halves (16B) global -> LDS
__device__ __forceinline__ void cp16(const _Float16* g, _Float16* l) {
#if USE_ASYNC
  __builtin_amdgcn_global_load_async_to_lds_b128((v4i*)g, (v4i*)l, 0, 0);
#else
  *(v8h*)l = *(const v8h*)g;
#endif
}
__device__ __forceinline__ void cp_join() {
#if USE_ASYNC
  __builtin_amdgcn_s_wait_asynccnt(0);
#endif
}

// WMMA f16 fragment from two aligned 8-half groups (ds_load_b128 x2)
__device__ __forceinline__ v16h frag16(const _Float16* p0, const _Float16* p1) {
  v8h a = *(const v8h*)p0;
  v8h b = *(const v8h*)p1;
  return __builtin_shufflevector(a, b, 0,1,2,3,4,5,6,7,8,9,10,11,12,13,14,15);
}
// A (16x32) fragment: row-major row pointer rp (16B aligned groups)
__device__ __forceinline__ v16h fragA(const _Float16* rp, bool hi) {
  return frag16(rp + (hi ? 8 : 0), rp + 16 + (hi ? 8 : 0));
}
// B (32x16) fragment: K-contiguous pointer per lane-column
__device__ __forceinline__ v16h fragB(const _Float16* kp, bool hi) {
  return frag16(kp + (hi ? 8 : 0), kp + 16 + (hi ? 8 : 0));
}

// ===========================================================================
// GEMM: C[M,N](f16) = scale * act( A[M,K](f16) @ Bt[N,K]^T + bias + addC )
// 128x64 block tile, 8 waves (32x32 each), K-step 32, exact tiling.
// ===========================================================================
__global__ __launch_bounds__(256) void st_gemm_kernel(
    const _Float16* __restrict__ A, int lda, long sA,
    const _Float16* __restrict__ Bt,              // [N][K] pre-transposed
    const float* __restrict__ bias,
    const _Float16* __restrict__ addC, long sAdd,
    _Float16* __restrict__ C, int ldc, long sC,
    int K, int act, float scaleOut)
{
  __shared__ _Float16 As[128][48];   // 96B rows (16B aligned groups)
  __shared__ _Float16 Bs[64][48];
  const int tid = threadIdx.x, lane = tid & 31, wave = tid >> 5;
  const int wm = wave >> 1, wn = wave & 1;
  const int bm = blockIdx.y * 128, bn = blockIdx.x * 64;
  const bool hi = lane >= 16;
  const int l = lane & 15;
  A += (long)blockIdx.z * sA;
  C += (long)blockIdx.z * sC;
  if (addC) addC += (long)blockIdx.z * sAdd;

  v8f acc[2][2] = {};
  for (int k0 = 0; k0 < K; k0 += 32) {
    // A tile 128x32: 512 16B segments, 2 per thread
#pragma unroll
    for (int it = 0; it < 2; ++it) {
      int op = tid + it * 256;
      int r = op >> 2, seg = op & 3;
      cp16(&A[(long)(bm + r) * lda + k0 + seg * 8], &As[r][seg * 8]);
    }
    // B tile 64x32 from Bt: 256 segments, 1 per thread
    {
      int r = tid >> 2, seg = tid & 3;
      cp16(&Bt[(long)(bn + r) * K + k0 + seg * 8], &Bs[r][seg * 8]);
    }
    if (k0 + 32 < K)
      __builtin_prefetch(&A[(long)(bm + (tid >> 1)) * lda + k0 + 32], 0, 1);
    cp_join();
    __syncthreads();

    v16h af[2], bf[2];
#pragma unroll
    for (int mi = 0; mi < 2; ++mi) af[mi] = fragA(&As[wm*32 + mi*16 + l][0], hi);
#pragma unroll
    for (int nj = 0; nj < 2; ++nj) bf[nj] = fragB(&Bs[wn*32 + nj*16 + l][0], hi);
#pragma unroll
    for (int mi = 0; mi < 2; ++mi)
#pragma unroll
      for (int nj = 0; nj < 2; ++nj)
        acc[mi][nj] = __builtin_amdgcn_wmma_f32_16x16x32_f16(
            false, af[mi], false, bf[nj], (short)0, acc[mi][nj], false, false);
    __syncthreads();
  }

#pragma unroll
  for (int mi = 0; mi < 2; ++mi)
#pragma unroll
    for (int nj = 0; nj < 2; ++nj)
#pragma unroll
      for (int g = 0; g < 8; ++g) {
        int r = bm + wm*32 + mi*16 + g + (hi ? 8 : 0);
        int c = bn + wn*32 + nj*16 + l;
        float v = acc[mi][nj][g];
        if (bias) v += bias[c];
        if (addC) v += (float)addC[(long)r * ldc + c];
        if (act == 1) {
          const float sl = 1.0507009873554805f, al = 1.6732632423543772f;
          v = sl * (v > 0.f ? v : al * (__expf(v) - 1.f));
        }
        v *= scaleOut;
        C[(long)r * ldc + c] = (_Float16)v;
      }
}

// ===========================================================================
// Encoder block-sparse attention (off-block logits are exactly 0):
//   Z_r = sum_blk exp(s) + (S-bs);  out = (exp(S)@Vblk + (Vtot-Vblk)) / Z
// Q pre-scaled by 1/sqrt(dh) in the conv epilogue.
// ===========================================================================
__global__ __launch_bounds__(256) void st_enc_attn_kernel(
    const _Float16* __restrict__ Q, const _Float16* __restrict__ K,
    const _Float16* __restrict__ V, const float* __restrict__ vsb,
    _Float16* __restrict__ O)
{
  const int blk = blockIdx.x, h = blockIdx.y, b = blockIdx.z;
  const int tid = threadIdx.x, lane = tid & 31, wave = tid >> 5;
  const bool hi = lane >= 16;
  const int l = lane & 15;
  const int wm = wave >> 1, wn = wave & 1;

  __shared__ char smem[52224];
  __shared__ float rowsum[128];
  __shared__ float vcorr[64];
  _Float16* Qs = (_Float16*)smem;             // [128][72]  phase A
  _Float16* Ks = (_Float16*)(smem + 18432);   // [128][72]  phase A
  _Float16* Ps = (_Float16*)smem;             // [128][136] phase B
  _Float16* Vt = (_Float16*)(smem + 34816);   // [64][136]  depth-major, phase B

  const size_t rowBase =
      ((size_t)b * HIST + (size_t)blk * 128) * D_MODEL + (size_t)h * DEPTH;

  // async stage Q,K (1024 16B segments each, 4 per thread)
#pragma unroll
  for (int it = 0; it < 4; ++it) {
    int op = tid + it * 256;
    int r = op >> 3, seg = op & 7;
    cp16(&Q[rowBase + (size_t)r * D_MODEL + seg * 8], &Qs[r * 72 + seg * 8]);
    cp16(&K[rowBase + (size_t)r * D_MODEL + seg * 8], &Ks[r * 72 + seg * 8]);
  }
  cp_join();
  __syncthreads();

  // scores 128x128: each wave 32 rows x 64 cols
  v8f sacc[2][4] = {};
#pragma unroll
  for (int ks = 0; ks < 2; ++ks) {
    v16h af[2], bf[4];
#pragma unroll
    for (int mi = 0; mi < 2; ++mi)
      af[mi] = fragA(&Qs[(wm*32 + mi*16 + l) * 72 + ks*32], hi);
#pragma unroll
    for (int nj = 0; nj < 4; ++nj)
      bf[nj] = fragB(&Ks[(wn*64 + nj*16 + l) * 72 + ks*32], hi);  // B = K^T
#pragma unroll
    for (int mi = 0; mi < 2; ++mi)
#pragma unroll
      for (int nj = 0; nj < 4; ++nj)
        sacc[mi][nj] = __builtin_amdgcn_wmma_f32_16x16x32_f16(
            false, af[mi], false, bf[nj], (short)0, sacc[mi][nj], false, false);
  }
  __syncthreads();   // Qs/Ks dead; LDS repurposed

  // exp -> P (f16, stride 136)
#pragma unroll
  for (int mi = 0; mi < 2; ++mi)
#pragma unroll
    for (int nj = 0; nj < 4; ++nj)
#pragma unroll
      for (int g = 0; g < 8; ++g) {
        int r = wm*32 + mi*16 + g + (hi ? 8 : 0);
        int c = wn*64 + nj*16 + l;
        Ps[r * 136 + c] = (_Float16)__expf(sacc[mi][nj][g]);
      }
  // stage V transposed (depth-major) for contiguous PV fragments
#pragma unroll
  for (int it = 0; it < 4; ++it) {
    int op = tid + it * 256;
    int r = op >> 3, sg = op & 7;
    v8h vv = *(const v8h*)&V[rowBase + (size_t)r * D_MODEL + sg * 8];
#pragma unroll
    for (int j = 0; j < 8; ++j) Vt[(sg*8 + j) * 136 + r] = vv[j];
  }
  __syncthreads();

  if (tid < 128) {
    float s = 0.f;
    for (int c = 0; c < 128; ++c) s += (float)Ps[tid * 136 + c];
    rowsum[tid] = s;
  } else if (tid < 192) {
    int d = tid - 128;
    const float* vp = vsb + ((size_t)(b * N_HEADS + h) * ENC_BLOCKS) * DEPTH;
    float tot = 0.f;
    for (int nb = 0; nb < ENC_BLOCKS; ++nb) tot += vp[nb * DEPTH + d];
    vcorr[d] = tot - vp[blk * DEPTH + d];
  }
  __syncthreads();

  // O = P @ V : 128x64, wave 32x32, K loop 128
  v8f oacc[2][2] = {};
#pragma unroll
  for (int ks = 0; ks < 4; ++ks) {
    v16h af[2], bf[2];
#pragma unroll
    for (int mi = 0; mi < 2; ++mi)
      af[mi] = fragA(&Ps[(wm*32 + mi*16 + l) * 136 + ks*32], hi);
#pragma unroll
    for (int nj = 0; nj < 2; ++nj)
      bf[nj] = fragB(&Vt[(wn*32 + nj*16 + l) * 136 + ks*32], hi);  // B = V (kv contig)
#pragma unroll
    for (int mi = 0; mi < 2; ++mi)
#pragma unroll
      for (int nj = 0; nj < 2; ++nj)
        oacc[mi][nj] = __builtin_amdgcn_wmma_f32_16x16x32_f16(
            false, af[mi], false, bf[nj], (short)0, oacc[mi][nj], false, false);
  }

#pragma unroll
  for (int mi = 0; mi < 2; ++mi)
#pragma unroll
    for (int nj = 0; nj < 2; ++nj)
#pragma unroll
      for (int g = 0; g < 8; ++g) {
        int r = wm*32 + mi*16 + g + (hi ? 8 : 0);
        int c = wn*32 + nj*16 + l;
        float Z = rowsum[r] + (float)(HIST - 128);
        O[rowBase + (size_t)r * D_MODEL + c] = (_Float16)((oacc[mi][nj][g] + vcorr[c]) / Z);
      }
}

// ===========================================================================
// Decoder attention, flash style: KV streamed in 64-row chunks, running
// denominator, optional causal mask.  Self (Skv=256) and cross (Skv=1024).
// Q pre-scaled by 1/sqrt(dh) in the conv epilogue.
// ===========================================================================
__global__ __launch_bounds__(256) void st_dec_attn_kernel(
    const _Float16* __restrict__ Q, const _Float16* __restrict__ K,
    const _Float16* __restrict__ V, _Float16* __restrict__ O,
    int Sq, int Skv, int causal)
{
  const int qb = blockIdx.x, h = blockIdx.y, b = blockIdx.z;
  const int tid = threadIdx.x, lane = tid & 31, wave = tid >> 5;
  const bool hi = lane >= 16;
  const int l = lane & 15;
  const int wm = wave >> 1, wn = wave & 1;

  __shared__ char smem[55296];
  __shared__ float Zrow[128];
  _Float16* Qs = (_Float16*)smem;             // [128][72]
  _Float16* Ks = (_Float16*)(smem + 18432);   // [64][72]
  _Float16* Vt = (_Float16*)(smem + 27648);   // [64][72] depth-major
  _Float16* Ps = (_Float16*)(smem + 36864);   // [128][72]

  const size_t qBase =
      ((size_t)b * Sq + (size_t)qb * 128) * D_MODEL + (size_t)h * DEPTH;
  const size_t kvBase = (size_t)b * Skv * D_MODEL + (size_t)h * DEPTH;

#pragma unroll
  for (int it = 0; it < 4; ++it) {
    int op = tid + it * 256;
    int r = op >> 3, seg = op & 7;
    cp16(&Q[qBase + (size_t)r * D_MODEL + seg * 8], &Qs[r * 72 + seg * 8]);
  }
  cp_join();
  if (tid < 128) Zrow[tid] = 0.f;

  v8f oacc[2][2] = {};
  const int nChunks = Skv / 64;
  for (int ch = 0; ch < nChunks; ++ch) {
    __syncthreads();   // previous readers done before restaging
    // K chunk async (512 segs, 2/thread); V chunk manual transposed
#pragma unroll
    for (int it = 0; it < 2; ++it) {
      int op = tid + it * 256;
      int r = op >> 3, sg = op & 7;
      size_t src = kvBase + (size_t)(ch * 64 + r) * D_MODEL + sg * 8;
      cp16(&K[src], &Ks[r * 72 + sg * 8]);
      v8h vv = *(const v8h*)&V[src];
#pragma unroll
      for (int j = 0; j < 8; ++j) Vt[(sg*8 + j) * 72 + r] = vv[j];
    }
    cp_join();
    __syncthreads();

    // scores chunk 128x64: wave 32x32
    v8f sacc[2][2] = {};
#pragma unroll
    for (int ks = 0; ks < 2; ++ks) {
      v16h af[2], bf[2];
#pragma unroll
      for (int mi = 0; mi < 2; ++mi)
        af[mi] = fragA(&Qs[(wm*32 + mi*16 + l) * 72 + ks*32], hi);
#pragma unroll
      for (int nj = 0; nj < 2; ++nj)
        bf[nj] = fragB(&Ks[(wn*32 + nj*16 + l) * 72 + ks*32], hi);
#pragma unroll
      for (int mi = 0; mi < 2; ++mi)
#pragma unroll
        for (int nj = 0; nj < 2; ++nj)
          sacc[mi][nj] = __builtin_amdgcn_wmma_f32_16x16x32_f16(
              false, af[mi], false, bf[nj], (short)0, sacc[mi][nj], false, false);
    }

    // mask + exp -> P
#pragma unroll
    for (int mi = 0; mi < 2; ++mi)
#pragma unroll
      for (int nj = 0; nj < 2; ++nj)
#pragma unroll
        for (int g = 0; g < 8; ++g) {
          int r = wm*32 + mi*16 + g + (hi ? 8 : 0);
          int c = wn*32 + nj*16 + l;
          int gq = qb * 128 + r, gk = ch * 64 + c;
          float p = (causal && gk > gq) ? 0.f : __expf(sacc[mi][nj][g]);
          Ps[r * 72 + c] = (_Float16)p;
        }
    __syncthreads();

    if (tid < 128) {
      float s = 0.f;
      for (int c = 0; c < 64; ++c) s += (float)Ps[tid * 72 + c];
      Zrow[tid] += s;
    }
    // O += P(128x64) @ V(64x64)
#pragma unroll
    for (int ks = 0; ks < 2; ++ks) {
      v16h af[2], bf[2];
#pragma unroll
      for (int mi = 0; mi < 2; ++mi)
        af[mi] = fragA(&Ps[(wm*32 + mi*16 + l) * 72 + ks*32], hi);
#pragma unroll
      for (int nj = 0; nj < 2; ++nj)
        bf[nj] = fragB(&Vt[(wn*32 + nj*16 + l) * 72 + ks*32], hi);
#pragma unroll
      for (int mi = 0; mi < 2; ++mi)
#pragma unroll
        for (int nj = 0; nj < 2; ++nj)
          oacc[mi][nj] = __builtin_amdgcn_wmma_f32_16x16x32_f16(
              false, af[mi], false, bf[nj], (short)0, oacc[mi][nj], false, false);
    }
  }
  __syncthreads();

#pragma unroll
  for (int mi = 0; mi < 2; ++mi)
#pragma unroll
    for (int nj = 0; nj < 2; ++nj)
#pragma unroll
      for (int g = 0; g < 8; ++g) {
        int r = wm*32 + mi*16 + g + (hi ? 8 : 0);
        int c = wn*32 + nj*16 + l;
        O[qBase + (size_t)r * D_MODEL + c] = (_Float16)(oacc[mi][nj][g] / Zrow[r]);
      }
}

// ===========================================================================
// helpers: converts, pad, V block sums, add+LN, final head
// ===========================================================================
__global__ void st_cvt_kernel(const float* __restrict__ in,
                              _Float16* __restrict__ out, long n)
{
  long i = (long)blockIdx.x * 256 + threadIdx.x;
  if (i < n) out[i] = (_Float16)in[i];
}

// f32 [K][N] -> f16 [N][K] (K,N multiples of 32)
__global__ __launch_bounds__(256) void st_cvt_t_kernel(
    const float* __restrict__ in, _Float16* __restrict__ out, int K, int N)
{
  __shared__ _Float16 t[32][33];
  int n0 = blockIdx.x * 32, k0 = blockIdx.y * 32;
  int tx = threadIdx.x & 31, ty = threadIdx.x >> 5;
#pragma unroll
  for (int j = 0; j < 32; j += 8)
    t[ty + j][tx] = (_Float16)in[(long)(k0 + ty + j) * N + n0 + tx];
  __syncthreads();
#pragma unroll
  for (int j = 0; j < 32; j += 8)
    out[(long)(n0 + ty + j) * K + k0 + tx] = t[tx][ty + j];
}

__global__ void st_pad_rows_kernel(const _Float16* __restrict__ x,
                                   _Float16* __restrict__ xp, int S, int C, int Pp)
{
  int b = blockIdx.y;
  int idx = blockIdx.x * 256 + threadIdx.x;
  int total = (S + Pp) * C;
  if (idx >= total) return;
  int r = idx / C, c = idx - r * C;
  _Float16 v = (r < Pp) ? (_Float16)0.f : x[((size_t)b * S + (r - Pp)) * C + c];
  xp[(size_t)b * (S + Pp) * C + idx] = v;
}

__global__ void st_vsum_kernel(const _Float16* __restrict__ V, float* __restrict__ vsb)
{
  int blk = blockIdx.x, h = blockIdx.y, b = blockIdx.z, d = threadIdx.x; // 64 thr
  float s = 0.f;
  for (int r = 0; r < 128; ++r)
    s += (float)V[((size_t)b * HIST + (size_t)blk * 128 + r) * D_MODEL + h * DEPTH + d];
  vsb[(((size_t)b * N_HEADS + h) * ENC_BLOCKS + blk) * DEPTH + d] = s;
}

__global__ __launch_bounds__(256) void st_add_ln_kernel(
    const _Float16* __restrict__ x, const _Float16* __restrict__ a,
    const float* __restrict__ g, const float* __restrict__ bt,
    _Float16* __restrict__ out)
{
  const int row = blockIdx.x, tid = threadIdx.x;
  __shared__ float red[256];
  const size_t base = (size_t)row * D_MODEL;
  float v0 = (float)x[base + tid] + (float)a[base + tid];
  float v1 = (float)x[base + 256 + tid] + (float)a[base + 256 + tid];
  red[tid] = v0 + v1;
  __syncthreads();
  for (int s = 128; s > 0; s >>= 1) { if (tid < s) red[tid] += red[tid + s]; __syncthreads(); }
  float mu = red[0] * (1.f / 512.f);
  __syncthreads();
  float d0 = v0 - mu, d1 = v1 - mu;
  red[tid] = d0 * d0 + d1 * d1;
  __syncthreads();
  for (int s = 128; s > 0; s >>= 1) { if (tid < s) red[tid] += red[tid + s]; __syncthreads(); }
  float rstd = rsqrtf(red[0] * (1.f / 512.f) + 1e-6f);
  out[base + tid]       = (_Float16)(d0 * rstd * g[tid]       + bt[tid]);
  out[base + 256 + tid] = (_Float16)(d1 * rstd * g[tid + 256] + bt[tid + 256]);
}

__global__ void st_final_kernel(const _Float16* __restrict__ y,
                                const float* __restrict__ w,
                                const float* __restrict__ b,
                                float* __restrict__ out, int rows)
{
  int wid = (blockIdx.x * 256 + threadIdx.x) >> 5;
  int lane = threadIdx.x & 31;
  if (wid >= rows) return;
  float s = 0.f;
  for (int i = lane; i < D_MODEL; i += 32)
    s += (float)y[(size_t)wid * D_MODEL + i] * w[i];
  for (int off = 16; off > 0; off >>= 1) s += __shfl_xor(s, off, 32);
  if (lane == 0) out[wid] = s + b[0];
}

// ===========================================================================
// Host orchestration
// ===========================================================================
extern "C" void kernel_launch(void* const* d_in, const int* in_sizes, int n_in,
                              void* d_out, int out_size, void* d_ws, size_t ws_size,
                              hipStream_t stream)
{
  (void)in_sizes; (void)n_in; (void)out_size; (void)ws_size;
  auto P = [&](int i) { return (const float*)d_in[i]; };
  const int EL = 6;            // enc layers: 16 tensors each
  const int DL = 6 + 2 * 16;   // dec layers: 26 tensors each

  char* base = (char*)d_ws;
  size_t off = 0;
  auto allocB = [&](size_t bytes) {
    size_t a = (off + 255) & ~(size_t)255;
    char* p = base + a; off = a + bytes; return p;
  };
  auto allocH = [&](size_t n) { return (_Float16*)allocB(n * 2); };
  auto allocF = [&](size_t n) { return (float*)allocB(n * 4); };

  auto cvt = [&](const float* src, size_t n) {
    _Float16* d = allocH(n);
    st_cvt_kernel<<<dim3((unsigned)((n + 255) / 256)), 256, 0, stream>>>(src, d, (long)n);
    return d;
  };
  auto cvtT = [&](const float* W, int K, int N) {   // -> f16 [N][K]
    _Float16* d = allocH((size_t)K * N);
    st_cvt_t_kernel<<<dim3(N / 32, K / 32), 256, 0, stream>>>(W, d, K, N);
    return (const _Float16*)d;
  };

  const size_t ME = (size_t)BATCH * HIST;   // 4096
  const size_t MD = (size_t)BATCH * FLEN;   // 1024
  _Float16* xe    = allocH(ME * D_MODEL);
  _Float16* xpad  = allocH((size_t)BATCH * (HIST + 2) * D_MODEL);
  _Float16* qb    = allocH(ME * D_MODEL);
  _Float16* kb    = allocH(ME * D_MODEL);
  _Float16* vb    = allocH(ME * D_MODEL);
  _Float16* attn  = allocH(ME * D_MODEL);
  _Float16* t1    = allocH(ME * D_MODEL);
  _Float16* o1    = allocH(ME * D_MODEL);
  _Float16* ffn_h = allocH(ME * DFF);
  float*    vsb   = allocF((size_t)BATCH * N_HEADS * ENC_BLOCKS * DEPTH);
  _Float16* yd    = allocH(MD * D_MODEL);
  _Float16* ypad  = allocH((size_t)BATCH * (FLEN + 2) * D_MODEL);
  _Float16* qd    = allocH(MD * D_MODEL);
  _Float16* kd    = allocH(MD * D_MODEL);
  _Float16* vd    = allocH(MD * D_MODEL);
  _Float16* attnd = allocH(MD * D_MODEL);
  _Float16* t1d   = allocH(MD * D_MODEL);
  _Float16* o1d   = allocH(MD * D_MODEL);
  _Float16* o2d   = allocH(MD * D_MODEL);
  _Float16* ffn_hd= allocH(MD * DFF);

  auto gemm = [&](const _Float16* A, int lda, long sA, const _Float16* Bt,
                  const float* bias, const _Float16* addC, long sAdd,
                  _Float16* C, int ldc, long sC,
                  int M, int N, int K, int act, float scale, int nb) {
    dim3 grid(N / 64, M / 128, nb);
    st_gemm_kernel<<<grid, 256, 0, stream>>>(A, lda, sA, Bt, bias, addC, sAdd,
                                             C, ldc, sC, K, act, scale);
  };
  const float QS = 0.125f;  // 1/sqrt(DEPTH), folded into q-conv epilogue

  // ---- inputs to f16 ----
  const _Float16* enc_x16 = cvt(P(0), ME * FEAT);
  const _Float16* dec_x16 = cvt(P(1), MD * FEAT);

  // ================= encoder =================
  gemm(enc_x16, FEAT, 0, cvtT(P(2), FEAT, D_MODEL), nullptr, nullptr, 0,
       xe, D_MODEL, 0, (int)ME, D_MODEL, FEAT, 0, 1.f, 1);

  for (int lyr = 0; lyr < 2; ++lyr) {
    int pb = EL + lyr * 16;
    const _Float16* q3t[3];
    for (int j = 0; j < 3; ++j)
      q3t[j] = cvtT(P(pb + 0) + (size_t)j * D_MODEL * D_MODEL, D_MODEL, D_MODEL);
    const _Float16* wkT = cvtT(P(pb + 2), D_MODEL, D_MODEL);
    const _Float16* wvT = cvtT(P(pb + 4), D_MODEL, D_MODEL);
    const _Float16* woT = cvtT(P(pb + 6), D_MODEL, D_MODEL);
    const _Float16* w1T = cvtT(P(pb + 8), D_MODEL, DFF);
    const _Float16* w2T = cvtT(P(pb + 10), DFF, D_MODEL);
    const float *wq_b = P(pb+1), *wk_b = P(pb+3), *wv_b = P(pb+5), *wo_b = P(pb+7),
                *fb1 = P(pb+9), *fb2 = P(pb+11),
                *l1g = P(pb+12), *l1b = P(pb+13), *l2g = P(pb+14), *l2b = P(pb+15);

    // causal conv (K=3) for Q, scale 1/sqrt(dh) folded into last tap
    {
      dim3 g(((HIST + 2) * D_MODEL + 255) / 256, BATCH);
      st_pad_rows_kernel<<<g, 256, 0, stream>>>(xe, xpad, HIST, D_MODEL, 2);
    }
    for (int j = 0; j < 3; ++j)
      gemm(xpad + (size_t)j * D_MODEL, D_MODEL, (long)(HIST + 2) * D_MODEL, q3t[j],
           j == 0 ? wq_b : nullptr, j == 0 ? nullptr : qb, (long)HIST * D_MODEL,
           qb, D_MODEL, (long)HIST * D_MODEL, HIST, D_MODEL, D_MODEL, 0,
           j == 2 ? QS : 1.f, BATCH);
    gemm(xe, D_MODEL, 0, wkT, wk_b, nullptr, 0, kb, D_MODEL, 0,
         (int)ME, D_MODEL, D_MODEL, 0, 1.f, 1);
    gemm(xe, D_MODEL, 0, wvT, wv_b, nullptr, 0, vb, D_MODEL, 0,
         (int)ME, D_MODEL, D_MODEL, 0, 1.f, 1);

    st_vsum_kernel<<<dim3(ENC_BLOCKS, N_HEADS, BATCH), 64, 0, stream>>>(vb, vsb);
    st_enc_attn_kernel<<<dim3(ENC_BLOCKS, N_HEADS, BATCH), 256, 0, stream>>>(
        qb, kb, vb, vsb, attn);

    gemm(attn, D_MODEL, 0, woT, wo_b, nullptr, 0, t1, D_MODEL, 0,
         (int)ME, D_MODEL, D_MODEL, 0, 1.f, 1);
    st_add_ln_kernel<<<(int)ME, 256, 0, stream>>>(xe, t1, l1g, l1b, o1);

    gemm(o1, D_MODEL, 0, w1T, fb1, nullptr, 0, ffn_h, DFF, 0,
         (int)ME, DFF, D_MODEL, 1, 1.f, 1);                         // fused SELU
    gemm(ffn_h, DFF, 0, w2T, fb2, nullptr, 0, t1, D_MODEL, 0,
         (int)ME, D_MODEL, DFF, 0, 1.f, 1);
    st_add_ln_kernel<<<(int)ME, 256, 0, stream>>>(o1, t1, l2g, l2b, xe);
  }
  // enc_out == xe

  // ================= decoder =================
  gemm(dec_x16, FEAT, 0, cvtT(P(3), FEAT, D_MODEL), nullptr, nullptr, 0,
       yd, D_MODEL, 0, (int)MD, D_MODEL, FEAT, 0, 1.f, 1);

  for (int lyr = 0; lyr < 2; ++lyr) {
    int pb = DL + lyr * 26;
    const _Float16 *q1t[3], *q2t[3];
    for (int j = 0; j < 3; ++j) {
      q1t[j] = cvtT(P(pb + 0) + (size_t)j * D_MODEL * D_MODEL, D_MODEL, D_MODEL);
      q2t[j] = cvtT(P(pb + 8) + (size_t)j * D_MODEL * D_MODEL, D_MODEL, D_MODEL);
    }
    const _Float16* k1T = cvtT(P(pb + 2),  D_MODEL, D_MODEL);
    const _Float16* v1T = cvtT(P(pb + 4),  D_MODEL, D_MODEL);
    const _Float16* o1T = cvtT(P(pb + 6),  D_MODEL, D_MODEL);
    const _Float16* k2T = cvtT(P(pb + 10), D_MODEL, D_MODEL);
    const _Float16* v2T = cvtT(P(pb + 12), D_MODEL, D_MODEL);
    const _Float16* o2T = cvtT(P(pb + 14), D_MODEL, D_MODEL);
    const _Float16* w1T = cvtT(P(pb + 16), D_MODEL, DFF);
    const _Float16* w2T = cvtT(P(pb + 18), DFF, D_MODEL);
    const float *q1b = P(pb+1), *k1b = P(pb+3), *v1b = P(pb+5), *o1b = P(pb+7),
                *q2b = P(pb+9), *k2b = P(pb+11), *v2b = P(pb+13), *o2b = P(pb+15),
                *fb1 = P(pb+17), *fb2 = P(pb+19),
                *l1g = P(pb+20), *l1b = P(pb+21), *l2g = P(pb+22), *l2b = P(pb+23),
                *l3g = P(pb+24), *l3b = P(pb+25);

    // --- masked self-attention ---
    {
      dim3 g(((FLEN + 2) * D_MODEL + 255) / 256, BATCH);
      st_pad_rows_kernel<<<g, 256, 0, stream>>>(yd, ypad, FLEN, D_MODEL, 2);
    }
    for (int j = 0; j < 3; ++j)
      gemm(ypad + (size_t)j * D_MODEL, D_MODEL, (long)(FLEN + 2) * D_MODEL, q1t[j],
           j == 0 ? q1b : nullptr, j == 0 ? nullptr : qd, (long)FLEN * D_MODEL,
           qd, D_MODEL, (long)FLEN * D_MODEL, FLEN, D_MODEL, D_MODEL, 0,
           j == 2 ? QS : 1.f, BATCH);
    gemm(yd, D_MODEL, 0, k1T, k1b, nullptr, 0, kd, D_MODEL, 0,
         (int)MD, D_MODEL, D_MODEL, 0, 1.f, 1);
    gemm(yd, D_MODEL, 0, v1T, v1b, nullptr, 0, vd, D_MODEL, 0,
         (int)MD, D_MODEL, D_MODEL, 0, 1.f, 1);
    st_dec_attn_kernel<<<dim3(FLEN / 128, N_HEADS, BATCH), 256, 0, stream>>>(
        qd, kd, vd, attnd, FLEN, FLEN, 1);
    gemm(attnd, D_MODEL, 0, o1T, o1b, nullptr, 0, t1d, D_MODEL, 0,
         (int)MD, D_MODEL, D_MODEL, 0, 1.f, 1);
    st_add_ln_kernel<<<(int)MD, 256, 0, stream>>>(yd, t1d, l1g, l1b, o1d);

    // --- cross attention (KV from encoder output) ---
    {
      dim3 g(((FLEN + 2) * D_MODEL + 255) / 256, BATCH);
      st_pad_rows_kernel<<<g, 256, 0, stream>>>(o1d, ypad, FLEN, D_MODEL, 2);
    }
    for (int j = 0; j < 3; ++j)
      gemm(ypad + (size_t)j * D_MODEL, D_MODEL, (long)(FLEN + 2) * D_MODEL, q2t[j],
           j == 0 ? q2b : nullptr, j == 0 ? nullptr : qd, (long)FLEN * D_MODEL,
           qd, D_MODEL, (long)FLEN * D_MODEL, FLEN, D_MODEL, D_MODEL, 0,
           j == 2 ? QS : 1.f, BATCH);
    gemm(xe, D_MODEL, 0, k2T, k2b, nullptr, 0, kb, D_MODEL, 0,
         (int)ME, D_MODEL, D_MODEL, 0, 1.f, 1);
    gemm(xe, D_MODEL, 0, v2T, v2b, nullptr, 0, vb, D_MODEL, 0,
         (int)ME, D_MODEL, D_MODEL, 0, 1.f, 1);
    st_dec_attn_kernel<<<dim3(FLEN / 128, N_HEADS, BATCH), 256, 0, stream>>>(
        qd, kb, vb, attnd, FLEN, HIST, 0);
    gemm(attnd, D_MODEL, 0, o2T, o2b, nullptr, 0, t1d, D_MODEL, 0,
         (int)MD, D_MODEL, D_MODEL, 0, 1.f, 1);
    st_add_ln_kernel<<<(int)MD, 256, 0, stream>>>(o1d, t1d, l2g, l2b, o2d);

    // --- FFN ---
    gemm(o2d, D_MODEL, 0, w1T, fb1, nullptr, 0, ffn_hd, DFF, 0,
         (int)MD, DFF, D_MODEL, 1, 1.f, 1);                          // fused SELU
    gemm(ffn_hd, DFF, 0, w2T, fb2, nullptr, 0, t1d, D_MODEL, 0,
         (int)MD, D_MODEL, DFF, 0, 1.f, 1);
    st_add_ln_kernel<<<(int)MD, 256, 0, stream>>>(o2d, t1d, l3g, l3b, yd);
  }

  // ================= final projection (512 -> 1, f32 out) =================
  st_final_kernel<<<((int)MD * 32 + 255) / 256, 256, 0, stream>>>(
      yd, P(4), P(5), (float*)d_out, (int)MD);
}
